// MultiHeadAttention_70557722738919
// MI455X (gfx1250) — matrix operand order repaired and checked by reading, not compiled, not run on hardware
//
#include <hip/hip_runtime.h>
#include <hip/hip_bf16.h>

// ---------------- problem constants ----------------
#define BATCH   4
#define SEQ     2048
#define DM      1024
#define NH      16
#define DHEAD   64
#define BSROWS  (BATCH * SEQ)   // 8192
#define LN_EPS  1e-5f

typedef __bf16 bf16;
typedef bf16  v16bf __attribute__((ext_vector_type(16)));
typedef float v8f   __attribute__((ext_vector_type(8)));

// ---------------- WMMA helpers (gfx1250, wave32) ----------------

__device__ __forceinline__ v8f zero8() {
    v8f z;
#pragma unroll
    for (int i = 0; i < 8; ++i) z[i] = 0.0f;
    return z;
}

// A-matrix 16x32 bf16 fragment, row-major source with leading dim `ld`.
// Layout (ISA 7.12.2): lane m = lane&15; lanes0-15 K in {0..7,16..23},
// lanes16-31 K in {8..15,24..31}; VGPR v holds K pair (2 elems).
__device__ __forceinline__ v16bf frag_a_ld(const bf16* p, int ld) {
    const int lane = threadIdx.x & 31;
    const int m  = lane & 15;
    const int kb = (lane & 16) ? 8 : 0;
    v16bf a;
#pragma unroll
    for (int v = 0; v < 8; ++v) {
        const int k = ((v & 4) << 2) + kb + 2 * (v & 3);  // +16 for v>=4
        a[2 * v]     = p[m * ld + k];
        a[2 * v + 1] = p[m * ld + k + 1];
    }
    return a;
}

// B-matrix 32x16 bf16 fragment. Source laid out [n][k] row-major, leading dim `ld`.
// Layout: lane n = lane&15; lanes0-15 K in {0..7,16..23}, lanes16-31 K in {8..15,24..31}.
__device__ __forceinline__ v16bf frag_b_ld(const bf16* p, int ld) {
    const int lane = threadIdx.x & 31;
    const int n  = lane & 15;
    const int kb = (lane & 16) ? 8 : 0;
    v16bf b;
#pragma unroll
    for (int v = 0; v < 8; ++v) {
        const int k = ((v & 4) << 2) + kb + 2 * (v & 3);
        b[2 * v]     = p[n * ld + k];
        b[2 * v + 1] = p[n * ld + k + 1];
    }
    return b;
}

__device__ __forceinline__ v8f wmma_bf16(v16bf a, v16bf b, v8f c) {
    return __builtin_amdgcn_wmma_f32_16x16x32_bf16(false, a, false, b,
                                                   (short)0, c, false, false);
}

// ---------------- async global->LDS copy (CDNA5 ASYNCcnt path) ----------------
// LDS offset = low 32 bits of the generic shared-aperture address
// (ISA 10.2).
__device__ __forceinline__ unsigned int lds_off32(const void* p) {
    return (unsigned int)(reinterpret_cast<uintptr_t>(p));
}
__device__ __forceinline__ void async_ld_b128(unsigned int lds, const void* gptr) {
    const unsigned long long ga = (unsigned long long)(uintptr_t)gptr;
    asm volatile("global_load_async_to_lds_b128 %0, %1, off"
                 :: "v"(lds), "v"(ga) : "memory");
}
__device__ __forceinline__ void wait_async0() {
    asm volatile("s_wait_asynccnt 0x0" ::: "memory");
}

// ---------------- Kernel 1: fused QKV projections ----------------
// C(8192x1024) = X(8192x1024) * W^T + b, X fp32 -> bf16 on the fly.
// Output layout (B, H, S, dh) bf16. q additionally scaled by 1/sqrt(dh).
__global__ __launch_bounds__(256) void qkv_proj_kernel(
    const float* __restrict__ Qi, const float* __restrict__ Ki, const float* __restrict__ Vi,
    const float* __restrict__ Wq, const float* __restrict__ bq,
    const float* __restrict__ Wk, const float* __restrict__ bk,
    const float* __restrict__ Wv, const float* __restrict__ bv,
    bf16* __restrict__ qb, bf16* __restrict__ kb, bf16* __restrict__ vb)
{
    constexpr int BM = 128, BN = 64, BK = 32, LD = BK + 2;
    __shared__ __align__(16) bf16 As[BM][LD];
    __shared__ __align__(16) bf16 Bs[BN][LD];

    const int pz = blockIdx.z;
    const float* X    = (pz == 0) ? Qi : (pz == 1) ? Ki : Vi;
    const float* W    = (pz == 0) ? Wq : (pz == 1) ? Wk : Wv;
    const float* bias = (pz == 0) ? bq : (pz == 1) ? bk : bv;
    bf16*        O    = (pz == 0) ? qb : (pz == 1) ? kb : vb;
    const float  oscale = (pz == 0) ? 0.125f : 1.0f;   // 1/sqrt(64) folded into q

    const int tid  = threadIdx.x;
    const int m0   = blockIdx.x * BM;
    const int n0   = blockIdx.y * BN;
    const int wave = tid >> 5;
    const int wm   = wave >> 1;     // 0..3
    const int wn   = wave & 1;      // 0..1

    v8f acc[2][2];
#pragma unroll
    for (int i = 0; i < 2; ++i)
#pragma unroll
        for (int j = 0; j < 2; ++j) acc[i][j] = zero8();

    for (int k0 = 0; k0 < DM; k0 += BK) {
        __syncthreads();
        // A tile: 128x32 fp32 -> bf16 ; 1024 float4, 4 per thread
#pragma unroll
        for (int i = 0; i < 4; ++i) {
            const int q  = tid + 256 * i;
            const int r  = q >> 3;
            const int c4 = (q & 7) * 4;
            const float4 f = *reinterpret_cast<const float4*>(
                &X[(size_t)(m0 + r) * DM + k0 + c4]);
            As[r][c4 + 0] = (bf16)f.x;
            As[r][c4 + 1] = (bf16)f.y;
            As[r][c4 + 2] = (bf16)f.z;
            As[r][c4 + 3] = (bf16)f.w;
        }
        // B tile: W rows n0..n0+63, cols k0..k0+31 (W row-major => B[k][n]=W[n][k])
#pragma unroll
        for (int i = 0; i < 2; ++i) {
            const int q  = tid + 256 * i;
            const int r  = q >> 3;
            const int c4 = (q & 7) * 4;
            const float4 f = *reinterpret_cast<const float4*>(
                &W[(size_t)(n0 + r) * DM + k0 + c4]);
            Bs[r][c4 + 0] = (bf16)f.x;
            Bs[r][c4 + 1] = (bf16)f.y;
            Bs[r][c4 + 2] = (bf16)f.z;
            Bs[r][c4 + 3] = (bf16)f.w;
        }
        if (k0 + BK < DM) {
            __builtin_prefetch(&X[(size_t)(m0 + (tid & 127)) * DM + k0 + BK], 0, 1);
            __builtin_prefetch(&W[(size_t)(n0 + (tid & 63)) * DM + k0 + BK], 0, 1);
        }
        __syncthreads();

        v16bf af0 = frag_a_ld(&As[wm * 32][0], LD);
        v16bf af1 = frag_a_ld(&As[wm * 32 + 16][0], LD);
        v16bf bf0 = frag_b_ld(&Bs[wn * 32][0], LD);
        v16bf bf1 = frag_b_ld(&Bs[wn * 32 + 16][0], LD);
        acc[0][0] = wmma_bf16(af0, bf0, acc[0][0]);
        acc[0][1] = wmma_bf16(af0, bf1, acc[0][1]);
        acc[1][0] = wmma_bf16(af1, bf0, acc[1][0]);
        acc[1][1] = wmma_bf16(af1, bf1, acc[1][1]);
    }

    // Epilogue: bias, scale, write bf16 in (B,H,S,dh)
    const int lane = tid & 31;
    const int n    = lane & 15;
    const int mb   = (lane & 16) ? 8 : 0;
#pragma unroll
    for (int ms = 0; ms < 2; ++ms)
#pragma unroll
        for (int ns = 0; ns < 2; ++ns)
#pragma unroll
            for (int r = 0; r < 8; ++r) {
                const int grow = m0 + wm * 32 + ms * 16 + mb + r;
                const int gcol = n0 + wn * 32 + ns * 16 + n;
                const float val = (acc[ms][ns][r] + bias[gcol]) * oscale;
                const int bi = grow >> 11;          // / SEQ
                const int s  = grow & (SEQ - 1);
                const int h  = gcol >> 6;           // / DHEAD
                const int d  = gcol & (DHEAD - 1);
                O[(((size_t)bi * NH + h) * SEQ + s) * DHEAD + d] = (bf16)val;
            }
}

// ---------------- Kernel 2: flash attention ----------------
// One block per (b, h, 64-row q tile). 4 waves; wave w owns q rows [w*16, w*16+16).
// K tile streamed with GLOBAL_LOAD_ASYNC_TO_LDS_B128; V transposed through VGPRs.
__global__ __launch_bounds__(128) void attn_kernel(
    const bf16* __restrict__ qb, const bf16* __restrict__ kb, const bf16* __restrict__ vb,
    const int* __restrict__ mask, bf16* __restrict__ ctxb)
{
    constexpr int KT  = 32;
    constexpr int KLD = DHEAD + 8;                     // 72 elems = 144 B (16B aligned rows)
    __shared__ __align__(16) bf16 Ks[KT][KLD];         // [key][d]
    __shared__ __align__(16) bf16 Vs[DHEAD][KT + 2];   // [d][key] (transposed)
    __shared__ __align__(16) bf16 Ps[4][16][34];       // per-wave P staging
    __shared__ int msk[KT];

    const int b  = blockIdx.z;
    const int h  = blockIdx.y;
    const int q0 = blockIdx.x * 64;
    const int tid  = threadIdx.x;
    const int wave = tid >> 5;
    const int lane = tid & 31;
    const int n    = lane & 15;

    // q fragments held in registers for the whole kernel (q pre-scaled by 1/8)
    const bf16* qbase = qb + (((size_t)b * NH + h) * SEQ + q0 + wave * 16) * DHEAD;
    const v16bf aq0 = frag_a_ld(qbase, DHEAD);       // d 0..31
    const v16bf aq1 = frag_a_ld(qbase + 32, DHEAD);  // d 32..63

    v8f o[4];
#pragma unroll
    for (int i = 0; i < 4; ++i) o[i] = zero8();
    float m_s[8], l_s[8];
#pragma unroll
    for (int r = 0; r < 8; ++r) { m_s[r] = -1e30f; l_s[r] = 0.0f; }

    const size_t kvbase = ((size_t)b * NH + h) * SEQ * DHEAD;
    const int prow = (lane & 16) ? 8 : 0;

    for (int kt = 0; kt < SEQ / KT; ++kt) {
        const int k0 = kt * KT;
        __syncthreads();

        // K tile: 32 rows x 128 B = 256 x 16B chunks, async straight into LDS
        const char* kgb = reinterpret_cast<const char*>(kb + kvbase + (size_t)k0 * DHEAD);
#pragma unroll
        for (int i = 0; i < 2; ++i) {
            const int j   = tid + 128 * i;   // 0..255
            const int key = j >> 3;
            const int ch  = j & 7;
            async_ld_b128(lds_off32(&Ks[key][0]) + ch * 16, kgb + j * 16);
        }
        // V tile: 1024 dwords through VGPRs, stored transposed [d][key]
        const unsigned int* vg =
            reinterpret_cast<const unsigned int*>(vb + kvbase + (size_t)k0 * DHEAD);
#pragma unroll
        for (int i = 0; i < 8; ++i) {
            const int j   = tid + 128 * i;   // 0..1023
            const int key = j >> 5;
            const int du  = j & 31;          // dword within row -> d = 2*du
            unsigned int uv = vg[j];
            const bf16* pv = reinterpret_cast<const bf16*>(&uv);
            Vs[2 * du][key]     = pv[0];
            Vs[2 * du + 1][key] = pv[1];
        }
        if (tid < KT) msk[tid] = mask[(size_t)b * SEQ + k0 + tid];
        wait_async0();
        __syncthreads();

        // scores (q already scaled): 16x32 per wave = 2 tiles x 2 K-chunks
        v8f sc0 = wmma_bf16(aq0, frag_b_ld(&Ks[0][0], KLD), zero8());
        sc0     = wmma_bf16(aq1, frag_b_ld(&Ks[0][32], KLD), sc0);
        v8f sc1 = wmma_bf16(aq0, frag_b_ld(&Ks[16][0], KLD), zero8());
        sc1     = wmma_bf16(aq1, frag_b_ld(&Ks[16][32], KLD), sc1);

        const float mb0 = (msk[n] == 0)      ? -1e9f : 0.0f;
        const float mb1 = (msk[16 + n] == 0) ? -1e9f : 0.0f;

#pragma unroll
        for (int r = 0; r < 8; ++r) {
            const float x0 = sc0[r] + mb0;
            const float x1 = sc1[r] + mb1;
            float mx = fmaxf(x0, x1);
#pragma unroll
            for (int off = 1; off < 16; off <<= 1)
                mx = fmaxf(mx, __shfl_xor(mx, off, 32));
            const float mnew = fmaxf(m_s[r], mx);
            const float corr = __expf(m_s[r] - mnew);
            const float p0 = __expf(x0 - mnew);
            const float p1 = __expf(x1 - mnew);
            float rs = p0 + p1;
#pragma unroll
            for (int off = 1; off < 16; off <<= 1)
                rs += __shfl_xor(rs, off, 32);
            l_s[r] = l_s[r] * corr + rs;
            m_s[r] = mnew;
#pragma unroll
            for (int nt = 0; nt < 4; ++nt) o[nt][r] *= corr;
            Ps[wave][prow + r][n]      = (bf16)p0;
            Ps[wave][prow + r][16 + n] = (bf16)p1;
        }

        // ctx += P(16x32) x V(32x64)
        const v16bf ap = frag_a_ld(&Ps[wave][0][0], 34);
#pragma unroll
        for (int nt = 0; nt < 4; ++nt)
            o[nt] = wmma_bf16(ap, frag_b_ld(&Vs[nt * 16][0], KT + 2), o[nt]);
    }

    // finalize: divide by softmax denominator, write merged-head (B,S,D) bf16
#pragma unroll
    for (int r = 0; r < 8; ++r) {
        const float inv = 1.0f / l_s[r];
        const int s = q0 + wave * 16 + prow + r;
#pragma unroll
        for (int nt = 0; nt < 4; ++nt) {
            const int col = h * DHEAD + nt * 16 + n;
            ctxb[((size_t)b * SEQ + s) * DM + col] = (bf16)(o[nt][r] * inv);
        }
    }
}

// ---------------- Kernel 3: output projection ----------------
// out(8192x1024) fp32 = ctx_bf16(8192x1024) x Wo^T + bo
// A tile (already bf16) loaded with async global->LDS; B tile converts fp32->bf16.
__global__ __launch_bounds__(256) void oproj_kernel(
    const bf16* __restrict__ ctx, const float* __restrict__ Wo,
    const float* __restrict__ bo, float* __restrict__ outp)
{
    constexpr int BM = 128, BN = 64, BK = 32;
    constexpr int LDA = BK + 8;   // 40 elems = 80 B rows (16B aligned)
    constexpr int LDB = BK + 2;
    __shared__ __align__(16) bf16 As[BM][LDA];
    __shared__ __align__(16) bf16 Bs[BN][LDB];

    const int tid  = threadIdx.x;
    const int m0   = blockIdx.x * BM;
    const int n0   = blockIdx.y * BN;
    const int wave = tid >> 5;
    const int wm   = wave >> 1;
    const int wn   = wave & 1;

    v8f acc[2][2];
#pragma unroll
    for (int i = 0; i < 2; ++i)
#pragma unroll
        for (int j = 0; j < 2; ++j) acc[i][j] = zero8();

    for (int k0 = 0; k0 < DM; k0 += BK) {
        __syncthreads();
        // A tile: 128 rows x 64 B = 512 x 16B chunks, async into LDS
#pragma unroll
        for (int i = 0; i < 2; ++i) {
            const int j  = tid + 256 * i;   // 0..511
            const int r  = j >> 2;
            const int ch = j & 3;
            const char* src =
                reinterpret_cast<const char*>(ctx + (size_t)(m0 + r) * DM + k0) + ch * 16;
            async_ld_b128(lds_off32(&As[r][0]) + ch * 16, src);
        }
        // B tile from Wo fp32 -> bf16
#pragma unroll
        for (int i = 0; i < 2; ++i) {
            const int q  = tid + 256 * i;
            const int r  = q >> 3;
            const int c4 = (q & 7) * 4;
            const float4 f = *reinterpret_cast<const float4*>(
                &Wo[(size_t)(n0 + r) * DM + k0 + c4]);
            Bs[r][c4 + 0] = (bf16)f.x;
            Bs[r][c4 + 1] = (bf16)f.y;
            Bs[r][c4 + 2] = (bf16)f.z;
            Bs[r][c4 + 3] = (bf16)f.w;
        }
        if (k0 + BK < DM)
            __builtin_prefetch(&Wo[(size_t)(n0 + (tid & 63)) * DM + k0 + BK], 0, 1);
        wait_async0();
        __syncthreads();

        v16bf af0 = frag_a_ld(&As[wm * 32][0], LDA);
        v16bf af1 = frag_a_ld(&As[wm * 32 + 16][0], LDA);
        v16bf bf0 = frag_b_ld(&Bs[wn * 32][0], LDB);
        v16bf bf1 = frag_b_ld(&Bs[wn * 32 + 16][0], LDB);
        acc[0][0] = wmma_bf16(af0, bf0, acc[0][0]);
        acc[0][1] = wmma_bf16(af0, bf1, acc[0][1]);
        acc[1][0] = wmma_bf16(af1, bf0, acc[1][0]);
        acc[1][1] = wmma_bf16(af1, bf1, acc[1][1]);
    }

    const int lane = tid & 31;
    const int n    = lane & 15;
    const int mb   = (lane & 16) ? 8 : 0;
#pragma unroll
    for (int ms = 0; ms < 2; ++ms)
#pragma unroll
        for (int ns = 0; ns < 2; ++ns)
#pragma unroll
            for (int r = 0; r < 8; ++r) {
                const int grow = m0 + wm * 32 + ms * 16 + mb + r;
                const int gcol = n0 + wn * 32 + ns * 16 + n;
                outp[(size_t)grow * DM + gcol] = acc[ms][ns][r] + bo[gcol];
            }
}

// ---------------- Kernel 4: residual + LayerNorm (in place on d_out) ----------------
__global__ __launch_bounds__(256) void ln_kernel(
    float* __restrict__ out, const float* __restrict__ Qin,
    const float* __restrict__ gamma, const float* __restrict__ beta)
{
    __shared__ float red[16];
    const int row = blockIdx.x;
    const int tid = threadIdx.x;
    const size_t base = (size_t)row * DM;
    const int c = tid * 4;

    const float4 o4 = *reinterpret_cast<const float4*>(&out[base + c]);
    const float4 q4 = *reinterpret_cast<const float4*>(&Qin[base + c]);
    const float x0 = o4.x + q4.x, x1 = o4.y + q4.y;
    const float x2 = o4.z + q4.z, x3 = o4.w + q4.w;

    float s  = x0 + x1 + x2 + x3;
    float ss = x0 * x0 + x1 * x1 + x2 * x2 + x3 * x3;
#pragma unroll
    for (int off = 1; off < 32; off <<= 1) {
        s  += __shfl_xor(s, off, 32);
        ss += __shfl_xor(ss, off, 32);
    }
    const int wave = tid >> 5, lane = tid & 31;
    if (lane == 0) { red[wave] = s; red[8 + wave] = ss; }
    __syncthreads();
    if (tid == 0) {
        float ts = 0.0f, tss = 0.0f;
#pragma unroll
        for (int w = 0; w < 8; ++w) { ts += red[w]; tss += red[8 + w]; }
        red[0] = ts; red[8] = tss;
    }
    __syncthreads();
    const float mu   = red[0] * (1.0f / DM);
    const float var  = red[8] * (1.0f / DM) - mu * mu;
    const float rinv = rsqrtf(var + LN_EPS);

    const float4 g4 = *reinterpret_cast<const float4*>(&gamma[c]);
    const float4 b4 = *reinterpret_cast<const float4*>(&beta[c]);
    float4 y;
    y.x = (x0 - mu) * rinv * g4.x + b4.x;
    y.y = (x1 - mu) * rinv * g4.y + b4.y;
    y.z = (x2 - mu) * rinv * g4.z + b4.z;
    y.w = (x3 - mu) * rinv * g4.w + b4.w;
    *reinterpret_cast<float4*>(&out[base + c]) = y;
}

// ---------------- host launcher ----------------
extern "C" void kernel_launch(void* const* d_in, const int* in_sizes, int n_in,
                              void* d_out, int out_size, void* d_ws, size_t ws_size,
                              hipStream_t stream) {
    const float* Q     = (const float*)d_in[0];
    const float* K     = (const float*)d_in[1];
    const float* V     = (const float*)d_in[2];
    const int*   mask  = (const int*)d_in[3];
    const float* Wq    = (const float*)d_in[4];
    const float* bq    = (const float*)d_in[5];
    const float* Wk    = (const float*)d_in[6];
    const float* bk    = (const float*)d_in[7];
    const float* Wv    = (const float*)d_in[8];
    const float* bv    = (const float*)d_in[9];
    const float* Wo    = (const float*)d_in[10];
    const float* bo    = (const float*)d_in[11];
    const float* gamma = (const float*)d_in[12];
    const float* beta  = (const float*)d_in[13];

    const size_t qkv_bytes = (size_t)BSROWS * DM * sizeof(bf16);  // 16 MB each
    char* ws = (char*)d_ws;
    bf16* qb   = (bf16*)(ws);
    bf16* kb   = (bf16*)(ws + qkv_bytes);
    bf16* vb   = (bf16*)(ws + 2 * qkv_bytes);
    bf16* ctxb = (bf16*)(ws + 3 * qkv_bytes);
    float* outp = (float*)d_out;

    dim3 g1(BSROWS / 128, DM / 64, 3);
    qkv_proj_kernel<<<g1, 256, 0, stream>>>(Q, K, V, Wq, bq, Wk, bk, Wv, bv, qb, kb, vb);

    dim3 g2(SEQ / 64, NH, BATCH);
    attn_kernel<<<g2, 128, 0, stream>>>(qb, kb, vb, mask, ctxb);

    dim3 g3(BSROWS / 128, DM / 64, 1);
    oproj_kernel<<<g3, 256, 0, stream>>>(ctxb, Wo, bo, outp);

    ln_kernel<<<BSROWS, 256, 0, stream>>>(outp, Q, gamma, beta);
}
